// Transformer_9878424781284
// MI455X (gfx1250) — matrix operand order repaired and checked by reading, not compiled, not run on hardware
//
#include <hip/hip_runtime.h>
#include <math.h>

// ---------------------------------------------------------------------------
// Transformer encoder-decoder forward for MI455X (gfx1250, wave32, WMMA).
// All dense math runs through one bf16 WMMA GEMM kernel (f32 accumulate),
// double-buffered LDS staging, fused alpha/bias/residual/relu epilogues.
// LN / softmax / embedding are small VALU kernels. Model + activations fit
// in the 192MB L2 -> matrix-throughput bound, so v_wmma_f32_16x16x32_bf16
// (8x the MACs/issue of the f32 WMMA) is the right instruction.
// ---------------------------------------------------------------------------

typedef __attribute__((ext_vector_type(16))) __bf16 v16bf;
typedef __attribute__((ext_vector_type(8)))  float  v8f;

union FragBF {
    v16bf v;
    unsigned u[8];
    unsigned short s[16];
};

__device__ __forceinline__ unsigned short f2bf(float f) {
    unsigned u = __builtin_bit_cast(unsigned, f);
    u += 0x7FFFu + ((u >> 16) & 1u);   // round-to-nearest-even
    return (unsigned short)(u >> 16);
}

__device__ __forceinline__ unsigned pack2bf(float a, float b) {
    return (unsigned)f2bf(a) | ((unsigned)f2bf(b) << 16);
}

// ---------------------------------------------------------------------------
// Tiled GEMM: C[M,N] = alpha * A[M,K] @ B[K,N] (+bias[N]) (+R[M,N]) (relu?)
// Block tile 128x128, K-step 32, 8 waves (2x4), each wave 64x32 via
// 4x2 v_wmma_f32_16x16x32_bf16 accumulators. Double-buffered LDS: global
// loads for slab k+1 issue before the WMMAs of slab k (one barrier/iter).
// Batched: z = outer*innerH + inner; each operand offset by its two strides.
// ---------------------------------------------------------------------------
__global__ __launch_bounds__(256) void gemm_bf16_wmma(
    const float* __restrict__ A, long long lda, long long sAo, long long sAi,
    const float* __restrict__ B, long long ldb, long long sBo, long long sBi,
    int transB,
    float* C, long long ldc, long long sCo, long long sCi,
    const float* __restrict__ bias,
    const float* R, long long ldr, long long sRo, long long sRi,
    int M, int N, int K, float alpha, int relu, int innerH)
{
    __shared__ unsigned short As[2][128][36];  // [buf][m][k], 72B rows (8B aligned)
    __shared__ unsigned short Bs[2][128][36];  // [buf][n][k] (n-major: K-pairs contiguous)

    const int tid   = threadIdx.x;
    const int lane  = tid & 31;
    const int wave  = tid >> 5;
    const int waveM = wave >> 2;               // 0..1
    const int waveN = wave & 3;                // 0..3
    const int blockM = blockIdx.y * 128;
    const int blockN = blockIdx.x * 128;

    const int bz    = blockIdx.z;
    const int outer = bz / innerH;
    const int inner = bz - outer * innerH;
    A += outer * sAo + inner * sAi;
    B += outer * sBo + inner * sBi;
    C += outer * sCo + inner * sCi;
    if (R) R += outer * sRo + inner * sRi;

    const bool mOK = (blockM + 128) <= M;
    const bool nOK = (blockN + 128) <= N;

    v8f zero;
#pragma unroll
    for (int r = 0; r < 8; ++r) zero[r] = 0.0f;
    v8f acc[4][2];
#pragma unroll
    for (int mf = 0; mf < 4; ++mf)
#pragma unroll
        for (int nf = 0; nf < 2; ++nf) acc[mf][nf] = zero;

    float fa[4][4];   // staged A slab (global -> regs -> LDS)
    float fb[4][4];   // staged B slab

    // ---- global -> regs: A slab at k-offset kt (uniform fast path when aligned)
    auto loadA = [&](int kt) {
        const bool fast = mOK && ((kt + 32) <= K);
#pragma unroll
        for (int u = 0; u < 4; ++u) {
            int unit = tid + u * 256;
            int row  = unit >> 3;
            int k4   = (unit & 7) << 2;
            int gm   = blockM + row;
            int gk   = kt + k4;
            if (fast) {
                const float* p = A + (long long)gm * lda + gk;
                float4 vv = *(const float4*)p;
                fa[u][0] = vv.x; fa[u][1] = vv.y; fa[u][2] = vv.z; fa[u][3] = vv.w;
                if (u == 0 && (kt + 96) <= K)
                    __builtin_prefetch(p + 64, 0, 0);    // global_prefetch_b8 (k+64 slab)
            } else {
                float f0 = 0.f, f1 = 0.f, f2 = 0.f, f3 = 0.f;
                if (gm < M) {
                    const float* p = A + (long long)gm * lda;
                    if (gk + 0 < K) f0 = p[gk + 0];
                    if (gk + 1 < K) f1 = p[gk + 1];
                    if (gk + 2 < K) f2 = p[gk + 2];
                    if (gk + 3 < K) f3 = p[gk + 3];
                }
                fa[u][0] = f0; fa[u][1] = f1; fa[u][2] = f2; fa[u][3] = f3;
            }
        }
    };

    // ---- global -> regs: B slab (layout depends on transB; both uniform-branch)
    auto loadB = [&](int kt) {
        if (transB) {      // B element (k,n) = Bp[n*ldb + k]
            const bool fast = nOK && ((kt + 32) <= K);
#pragma unroll
            for (int u = 0; u < 4; ++u) {
                int unit = tid + u * 256;
                int n    = unit >> 3;
                int k4   = (unit & 7) << 2;
                int gn   = blockN + n;
                int gk   = kt + k4;
                if (fast) {
                    const float* p = B + (long long)gn * ldb + gk;
                    float4 vv = *(const float4*)p;
                    fb[u][0] = vv.x; fb[u][1] = vv.y; fb[u][2] = vv.z; fb[u][3] = vv.w;
                    if (u == 0 && (kt + 96) <= K)
                        __builtin_prefetch(p + 64, 0, 0);
                } else {
                    float f0 = 0.f, f1 = 0.f, f2 = 0.f, f3 = 0.f;
                    if (gn < N) {
                        const float* p = B + (long long)gn * ldb;
                        if (gk + 0 < K) f0 = p[gk + 0];
                        if (gk + 1 < K) f1 = p[gk + 1];
                        if (gk + 2 < K) f2 = p[gk + 2];
                        if (gk + 3 < K) f3 = p[gk + 3];
                    }
                    fb[u][0] = f0; fb[u][1] = f1; fb[u][2] = f2; fb[u][3] = f3;
                }
            }
        } else {           // B element (k,n) = Bp[k*ldb + n]
            const bool fast = nOK && ((kt + 32) <= K);
#pragma unroll
            for (int u = 0; u < 4; ++u) {
                int unit = tid + u * 256;
                int k    = unit >> 5;             // 0..31
                int n4   = (unit & 31) << 2;      // 0..124
                int gk   = kt + k;
                int gn   = blockN + n4;
                if (fast) {
                    const float* p = B + (long long)gk * ldb + gn;
                    float4 vv = *(const float4*)p;
                    fb[u][0] = vv.x; fb[u][1] = vv.y; fb[u][2] = vv.z; fb[u][3] = vv.w;
                    if (u == 0 && (kt + 96) <= K)
                        __builtin_prefetch(p + 64 * ldb, 0, 0);
                } else {
                    float f0 = 0.f, f1 = 0.f, f2 = 0.f, f3 = 0.f;
                    if (gk < K) {
                        const float* p = B + (long long)gk * ldb;
                        if (gn + 0 < N) f0 = p[gn + 0];
                        if (gn + 1 < N) f1 = p[gn + 1];
                        if (gn + 2 < N) f2 = p[gn + 2];
                        if (gn + 3 < N) f3 = p[gn + 3];
                    }
                    fb[u][0] = f0; fb[u][1] = f1; fb[u][2] = f2; fb[u][3] = f3;
                }
            }
        }
    };

    // ---- regs -> LDS (bf16 convert on the way in)
    auto storeA = [&](int buf) {
#pragma unroll
        for (int u = 0; u < 4; ++u) {
            int unit = tid + u * 256;
            int row  = unit >> 3;
            int k4   = (unit & 7) << 2;
            *(unsigned*)&As[buf][row][k4]     = pack2bf(fa[u][0], fa[u][1]);
            *(unsigned*)&As[buf][row][k4 + 2] = pack2bf(fa[u][2], fa[u][3]);
        }
    };
    auto storeB = [&](int buf) {
        if (transB) {
#pragma unroll
            for (int u = 0; u < 4; ++u) {
                int unit = tid + u * 256;
                int n    = unit >> 3;
                int k4   = (unit & 7) << 2;
                *(unsigned*)&Bs[buf][n][k4]     = pack2bf(fb[u][0], fb[u][1]);
                *(unsigned*)&Bs[buf][n][k4 + 2] = pack2bf(fb[u][2], fb[u][3]);
            }
        } else {
#pragma unroll
            for (int u = 0; u < 4; ++u) {
                int unit = tid + u * 256;
                int k    = unit >> 5;
                int n4   = (unit & 31) << 2;
                Bs[buf][n4 + 0][k] = f2bf(fb[u][0]);
                Bs[buf][n4 + 1][k] = f2bf(fb[u][1]);
                Bs[buf][n4 + 2][k] = f2bf(fb[u][2]);
                Bs[buf][n4 + 3][k] = f2bf(fb[u][3]);
            }
        }
    };

    // ---- prologue: stage slab 0 into buffer 0
    loadA(0); loadB(0);
    storeA(0); storeB(0);
    __syncthreads();

    int cur = 0;
    for (int kt = 0; kt < K; kt += 32) {
        const bool more = (kt + 32) < K;
        if (more) { loadA(kt + 32); loadB(kt + 32); }   // global loads overlap WMMA

        // ---- fragments per ISA 7.12.2 layouts (wave32), from buffer `cur`
        FragBF afr[4];
#pragma unroll
        for (int mf = 0; mf < 4; ++mf) {
            int mrow  = waveM * 64 + mf * 16 + (lane & 15);
            int khalf = (lane < 16) ? 0 : 8;
#pragma unroll
            for (int j = 0; j < 8; ++j) {
                int kk = ((j < 4) ? (2 * j) : (16 + 2 * (j - 4))) + khalf;
                afr[mf].u[j] = *(const unsigned*)&As[cur][mrow][kk];
            }
        }
        FragBF bfr[2];
#pragma unroll
        for (int nf = 0; nf < 2; ++nf) {
            int ncol = waveN * 32 + nf * 16 + (lane & 15);
            int kh   = (lane < 16) ? 0 : 8;
#pragma unroll
            for (int j = 0; j < 8; ++j)
                bfr[nf].u[j] = *(const unsigned*)&Bs[cur][ncol][((j < 4) ? (2 * j) : (16 + 2 * (j - 4))) + kh];
        }

        // ---- 8 WMMAs per wave per K-slab
#pragma unroll
        for (int mf = 0; mf < 4; ++mf)
#pragma unroll
            for (int nf = 0; nf < 2; ++nf)
                acc[mf][nf] = __builtin_amdgcn_wmma_f32_16x16x32_bf16(
                    false, afr[mf].v, false, bfr[nf].v,
                    (short)0, acc[mf][nf], false, false);

        if (more) {
            storeA(cur ^ 1); storeB(cur ^ 1);           // fill the other buffer
            __syncthreads();                            // one barrier per K-slab
            cur ^= 1;
        }
    }

    // ---- epilogue: alpha, bias, residual, relu, store (C layout per 7.12.2)
    if (mOK && nOK) {
#pragma unroll
        for (int mf = 0; mf < 4; ++mf) {
#pragma unroll
            for (int nf = 0; nf < 2; ++nf) {
                int n     = blockN + waveN * 32 + nf * 16 + (lane & 15);
                int mbase = blockM + waveM * 64 + mf * 16 + ((lane < 16) ? 0 : 8);
#pragma unroll
                for (int r = 0; r < 8; ++r) {
                    int m = mbase + r;
                    float val = alpha * acc[mf][nf][r];
                    if (bias) val += bias[n];
                    if (R)    val += R[(long long)m * ldr + n];
                    if (relu) val = fmaxf(val, 0.0f);
                    C[(long long)m * ldc + n] = val;
                }
            }
        }
    } else {
#pragma unroll
        for (int mf = 0; mf < 4; ++mf) {
#pragma unroll
            for (int nf = 0; nf < 2; ++nf) {
                int n     = blockN + waveN * 32 + nf * 16 + (lane & 15);
                int mbase = blockM + waveM * 64 + mf * 16 + ((lane < 16) ? 0 : 8);
#pragma unroll
                for (int r = 0; r < 8; ++r) {
                    int m = mbase + r;
                    if (m < M && n < N) {
                        float val = alpha * acc[mf][nf][r];
                        if (bias) val += bias[n];
                        if (R)    val += R[(long long)m * ldr + n];
                        if (relu) val = fmaxf(val, 0.0f);
                        C[(long long)m * ldc + n] = val;
                    }
                }
            }
        }
    }
}

// ---------------------------------------------------------------------------
// Row LayerNorm over D=512. One block per row, 256 threads, 2 elems/thread.
// ---------------------------------------------------------------------------
__global__ __launch_bounds__(256) void layernorm_kernel(
    const float* __restrict__ x, float* __restrict__ y,
    const float* __restrict__ scale, const float* __restrict__ bias)
{
    __shared__ float red[256];
    const long long row = blockIdx.x;
    const int tid = threadIdx.x;
    const float* xr = x + row * 512;
    float a = xr[tid];
    float b = xr[tid + 256];

    red[tid] = a + b;
    __syncthreads();
    for (int s = 128; s > 0; s >>= 1) {
        if (tid < s) red[tid] += red[tid + s];
        __syncthreads();
    }
    float mean = red[0] * (1.0f / 512.0f);
    __syncthreads();

    float da = a - mean, db = b - mean;
    red[tid] = da * da + db * db;
    __syncthreads();
    for (int s = 128; s > 0; s >>= 1) {
        if (tid < s) red[tid] += red[tid + s];
        __syncthreads();
    }
    float var = red[0] * (1.0f / 512.0f);
    float rs  = rsqrtf(var + 1e-5f);

    y[row * 512 + tid]       = da * rs * scale[tid]       + bias[tid];
    y[row * 512 + tid + 256] = db * rs * scale[tid + 256] + bias[tid + 256];
}

// ---------------------------------------------------------------------------
// Masked softmax over rows of scores[BH, Sq, Sk], Sk = 512.
// mode 0: key padding (tok = src ids, allowed = tok[b,k] != 0)
// mode 1: causal + query padding (tok = tgt ids, allowed = tok[b,q]!=0 && k<=q)
// ---------------------------------------------------------------------------
__global__ __launch_bounds__(256) void masked_softmax_kernel(
    float* scores, const int* __restrict__ tok,
    int mode, int Sq, int Sk, int Hh)
{
    __shared__ float red[256];
    const int q  = blockIdx.x;
    const int bh = blockIdx.y;
    const int b  = bh / Hh;
    const int tid = threadIdx.x;
    float* row = scores + ((long long)bh * Sq + q) * Sk;

    const int k0 = tid, k1 = tid + 256;
    bool a0, a1;
    if (mode == 0) {
        a0 = tok[b * Sk + k0] != 0;
        a1 = tok[b * Sk + k1] != 0;
    } else {
        bool qa = tok[b * Sq + q] != 0;
        a0 = qa && (k0 <= q);
        a1 = qa && (k1 <= q);
    }
    float v0 = a0 ? row[k0] : -1e9f;
    float v1 = a1 ? row[k1] : -1e9f;

    red[tid] = fmaxf(v0, v1);
    __syncthreads();
    for (int s = 128; s > 0; s >>= 1) {
        if (tid < s) red[tid] = fmaxf(red[tid], red[tid + s]);
        __syncthreads();
    }
    float mx = red[0];
    __syncthreads();

    float e0 = __expf(v0 - mx);
    float e1 = __expf(v1 - mx);
    red[tid] = e0 + e1;
    __syncthreads();
    for (int s = 128; s > 0; s >>= 1) {
        if (tid < s) red[tid] += red[tid + s];
        __syncthreads();
    }
    float inv = 1.0f / red[0];
    row[k0] = e0 * inv;
    row[k1] = e1 * inv;
}

// ---------------------------------------------------------------------------
// Embedding * sqrt(D) + sinusoidal positional encoding (computed in-kernel).
// ---------------------------------------------------------------------------
__global__ __launch_bounds__(256) void embed_kernel(
    const int* __restrict__ tok, const float* __restrict__ emb,
    float* __restrict__ out, int S)
{
    const int b   = blockIdx.y;
    const int pos = blockIdx.x;
    const int tid = threadIdx.x;
    const int t   = tok[b * S + pos];
#pragma unroll
    for (int e = 0; e < 2; ++e) {
        int d = tid + e * 256;
        float i2  = (float)(d & ~1);
        float ang = (float)pos * __expf(i2 * (-9.210340371976184f / 512.0f));
        float pe  = (d & 1) ? __cosf(ang) : __sinf(ang);
        out[((long long)(b * S + pos)) * 512 + d] =
            emb[(long long)t * 512 + d] * 22.62741699796952f + pe;
    }
}

// ---------------------------------------------------------------------------
// Host orchestration
// ---------------------------------------------------------------------------
static void run_gemm(hipStream_t st,
    const float* A, long long lda, long long sAo, long long sAi,
    const float* B, long long ldb, long long sBo, long long sBi, int transB,
    float* C, long long ldc, long long sCo, long long sCi,
    const float* bias,
    const float* R, long long ldr, long long sRo, long long sRi,
    int M, int N, int K, float alpha, int relu, int batches, int innerH)
{
    dim3 grid((N + 127) / 128, (M + 127) / 128, batches);
    gemm_bf16_wmma<<<grid, 256, 0, st>>>(
        A, lda, sAo, sAi, B, ldb, sBo, sBi, transB,
        C, ldc, sCo, sCi, bias, R, ldr, sRo, sRi,
        M, N, K, alpha, relu, innerH);
}

extern "C" void kernel_launch(void* const* d_in, const int* in_sizes, int n_in,
                              void* d_out, int out_size, void* d_ws, size_t ws_size,
                              hipStream_t stream)
{
    (void)in_sizes; (void)n_in; (void)out_size; (void)ws_size;

    const int*   src        = (const int*)d_in[0];
    const int*   tgt        = (const int*)d_in[1];
    const float* src_emb    = (const float*)d_in[2];
    const float* tgt_emb    = (const float*)d_in[3];
    const float* enc_attn_w = (const float*)d_in[4];
    const float* enc_attn_b = (const float*)d_in[5];
    const float* enc_ln     = (const float*)d_in[6];
    const float* enc_ff_w1  = (const float*)d_in[7];
    const float* enc_ff_b1  = (const float*)d_in[8];
    const float* enc_ff_w2  = (const float*)d_in[9];
    const float* enc_ff_b2  = (const float*)d_in[10];
    const float* enc_norm   = (const float*)d_in[11];
    const float* dec_attn_w = (const float*)d_in[12];
    const float* dec_attn_b = (const float*)d_in[13];
    const float* dec_ln     = (const float*)d_in[14];
    const float* dec_ff_w1  = (const float*)d_in[15];
    const float* dec_ff_b1  = (const float*)d_in[16];
    const float* dec_ff_w2  = (const float*)d_in[17];
    const float* dec_ff_b2  = (const float*)d_in[18];
    const float* dec_norm   = (const float*)d_in[19];
    const float* gen_w      = (const float*)d_in[20];
    const float* gen_b      = (const float*)d_in[21];

    constexpr long long Dd = 512, Ff = 2048, Vv = 16000, Bb = 8, Ss = 512,
                        Hh = 8, DK = 64;
    constexpr long long BS  = Bb * Ss;      // 4096 rows
    constexpr long long BSD = BS * Dd;
    constexpr long long BSF = BS * Ff;

    float* ws   = (float*)d_ws;
    float* x    = ws;                 // encoder state      [BS, D]
    float* h    = x    + BSD;         // LN output          [BS, D]
    float* qb   = h    + BSD;         // Q                  [BS, D]
    float* kb   = qb   + BSD;         // K                  [BS, D]
    float* vb   = kb   + BSD;         // V                  [BS, D]
    float* attn = vb   + BSD;         // attention output   [BS, D]
    float* mem  = attn + BSD;         // encoder memory     [BS, D]
    float* yb   = mem  + BSD;         // decoder state      [BS, D]
    float* ff   = yb   + BSD;         // FFN intermediate   [BS, F]
    float* sc   = ff   + BSF;         // scores             [B*H, S, S]

    const int MB = (int)BS;

    // =========================== Encoder ===========================
    embed_kernel<<<dim3((int)Ss, (int)Bb), 256, 0, stream>>>(src, src_emb, x, (int)Ss);

    for (int i = 0; i < 6; ++i) {
        const float* W    = enc_attn_w + (long long)i * 4 * Dd * Dd;
        const float* Bi   = enc_attn_b + (long long)i * 4 * Dd;
        const float* ln0s = enc_ln + ((long long)(i * 2 + 0) * 2 + 0) * Dd;
        const float* ln0b = ln0s + Dd;
        const float* ln1s = enc_ln + ((long long)(i * 2 + 1) * 2 + 0) * Dd;
        const float* ln1b = ln1s + Dd;

        layernorm_kernel<<<MB, 256, 0, stream>>>(x, h, ln0s, ln0b);
        run_gemm(stream, h, Dd,0,0, W + 0*Dd*Dd, Dd,0,0,0, qb, Dd,0,0, Bi + 0*Dd, nullptr,0,0,0, MB,(int)Dd,(int)Dd, 1.f,0, 1,1);
        run_gemm(stream, h, Dd,0,0, W + 1*Dd*Dd, Dd,0,0,0, kb, Dd,0,0, Bi + 1*Dd, nullptr,0,0,0, MB,(int)Dd,(int)Dd, 1.f,0, 1,1);
        run_gemm(stream, h, Dd,0,0, W + 2*Dd*Dd, Dd,0,0,0, vb, Dd,0,0, Bi + 2*Dd, nullptr,0,0,0, MB,(int)Dd,(int)Dd, 1.f,0, 1,1);
        // scores = Q @ K^T / 8   (batched over B*H, head offset = inner stride)
        run_gemm(stream, qb, Dd, Ss*Dd, DK, kb, Dd, Ss*Dd, DK, 1,
                 sc, Ss, Hh*Ss*Ss, Ss*Ss, nullptr, nullptr,0,0,0,
                 (int)Ss,(int)Ss,(int)DK, 0.125f,0, (int)(Bb*Hh),(int)Hh);
        masked_softmax_kernel<<<dim3((int)Ss,(int)(Bb*Hh)), 256, 0, stream>>>(sc, src, 0, (int)Ss, (int)Ss, (int)Hh);
        // attn = P @ V (head-interleaved write)
        run_gemm(stream, sc, Ss, Hh*Ss*Ss, Ss*Ss, vb, Dd, Ss*Dd, DK, 0,
                 attn, Dd, Ss*Dd, DK, nullptr, nullptr,0,0,0,
                 (int)Ss,(int)DK,(int)Ss, 1.f,0, (int)(Bb*Hh),(int)Hh);
        // x = attn @ Wo + bo + x
        run_gemm(stream, attn, Dd,0,0, W + 3*Dd*Dd, Dd,0,0,0, x, Dd,0,0, Bi + 3*Dd, x, Dd,0,0, MB,(int)Dd,(int)Dd, 1.f,0, 1,1);
        // FFN
        layernorm_kernel<<<MB, 256, 0, stream>>>(x, h, ln1s, ln1b);
        run_gemm(stream, h, Dd,0,0, enc_ff_w1 + (long long)i*Dd*Ff, Ff,0,0,0, ff, Ff,0,0, enc_ff_b1 + (long long)i*Ff, nullptr,0,0,0, MB,(int)Ff,(int)Dd, 1.f,1, 1,1);
        run_gemm(stream, ff, Ff,0,0, enc_ff_w2 + (long long)i*Ff*Dd, Dd,0,0,0, x, Dd,0,0, enc_ff_b2 + (long long)i*Dd, x, Dd,0,0, MB,(int)Dd,(int)Ff, 1.f,0, 1,1);
    }
    layernorm_kernel<<<MB, 256, 0, stream>>>(x, mem, enc_norm, enc_norm + Dd);

    // =========================== Decoder ===========================
    embed_kernel<<<dim3((int)Ss, (int)Bb), 256, 0, stream>>>(tgt, tgt_emb, yb, (int)Ss);

    for (int i = 0; i < 6; ++i) {
        const float* W    = dec_attn_w + (long long)i * 8 * Dd * Dd;
        const float* Bi   = dec_attn_b + (long long)i * 8 * Dd;
        const float* ln0s = dec_ln + ((long long)(i * 3 + 0) * 2 + 0) * Dd;
        const float* ln0b = ln0s + Dd;
        const float* ln1s = dec_ln + ((long long)(i * 3 + 1) * 2 + 0) * Dd;
        const float* ln1b = ln1s + Dd;
        const float* ln2s = dec_ln + ((long long)(i * 3 + 2) * 2 + 0) * Dd;
        const float* ln2b = ln2s + Dd;

        // ---- masked self-attention
        layernorm_kernel<<<MB, 256, 0, stream>>>(yb, h, ln0s, ln0b);
        run_gemm(stream, h, Dd,0,0, W + 0*Dd*Dd, Dd,0,0,0, qb, Dd,0,0, Bi + 0*Dd, nullptr,0,0,0, MB,(int)Dd,(int)Dd, 1.f,0, 1,1);
        run_gemm(stream, h, Dd,0,0, W + 1*Dd*Dd, Dd,0,0,0, kb, Dd,0,0, Bi + 1*Dd, nullptr,0,0,0, MB,(int)Dd,(int)Dd, 1.f,0, 1,1);
        run_gemm(stream, h, Dd,0,0, W + 2*Dd*Dd, Dd,0,0,0, vb, Dd,0,0, Bi + 2*Dd, nullptr,0,0,0, MB,(int)Dd,(int)Dd, 1.f,0, 1,1);
        run_gemm(stream, qb, Dd, Ss*Dd, DK, kb, Dd, Ss*Dd, DK, 1,
                 sc, Ss, Hh*Ss*Ss, Ss*Ss, nullptr, nullptr,0,0,0,
                 (int)Ss,(int)Ss,(int)DK, 0.125f,0, (int)(Bb*Hh),(int)Hh);
        masked_softmax_kernel<<<dim3((int)Ss,(int)(Bb*Hh)), 256, 0, stream>>>(sc, tgt, 1, (int)Ss, (int)Ss, (int)Hh);
        run_gemm(stream, sc, Ss, Hh*Ss*Ss, Ss*Ss, vb, Dd, Ss*Dd, DK, 0,
                 attn, Dd, Ss*Dd, DK, nullptr, nullptr,0,0,0,
                 (int)Ss,(int)DK,(int)Ss, 1.f,0, (int)(Bb*Hh),(int)Hh);
        run_gemm(stream, attn, Dd,0,0, W + 3*Dd*Dd, Dd,0,0,0, yb, Dd,0,0, Bi + 3*Dd, yb, Dd,0,0, MB,(int)Dd,(int)Dd, 1.f,0, 1,1);

        // ---- cross-attention (K,V from encoder memory)
        layernorm_kernel<<<MB, 256, 0, stream>>>(yb, h, ln1s, ln1b);
        run_gemm(stream, h,   Dd,0,0, W + 4*Dd*Dd, Dd,0,0,0, qb, Dd,0,0, Bi + 4*Dd, nullptr,0,0,0, MB,(int)Dd,(int)Dd, 1.f,0, 1,1);
        run_gemm(stream, mem, Dd,0,0, W + 5*Dd*Dd, Dd,0,0,0, kb, Dd,0,0, Bi + 5*Dd, nullptr,0,0,0, MB,(int)Dd,(int)Dd, 1.f,0, 1,1);
        run_gemm(stream, mem, Dd,0,0, W + 6*Dd*Dd, Dd,0,0,0, vb, Dd,0,0, Bi + 6*Dd, nullptr,0,0,0, MB,(int)Dd,(int)Dd, 1.f,0, 1,1);
        run_gemm(stream, qb, Dd, Ss*Dd, DK, kb, Dd, Ss*Dd, DK, 1,
                 sc, Ss, Hh*Ss*Ss, Ss*Ss, nullptr, nullptr,0,0,0,
                 (int)Ss,(int)Ss,(int)DK, 0.125f,0, (int)(Bb*Hh),(int)Hh);
        masked_softmax_kernel<<<dim3((int)Ss,(int)(Bb*Hh)), 256, 0, stream>>>(sc, src, 0, (int)Ss, (int)Ss, (int)Hh);
        run_gemm(stream, sc, Ss, Hh*Ss*Ss, Ss*Ss, vb, Dd, Ss*Dd, DK, 0,
                 attn, Dd, Ss*Dd, DK, nullptr, nullptr,0,0,0,
                 (int)Ss,(int)DK,(int)Ss, 1.f,0, (int)(Bb*Hh),(int)Hh);
        run_gemm(stream, attn, Dd,0,0, W + 7*Dd*Dd, Dd,0,0,0, yb, Dd,0,0, Bi + 7*Dd, yb, Dd,0,0, MB,(int)Dd,(int)Dd, 1.f,0, 1,1);

        // ---- FFN
        layernorm_kernel<<<MB, 256, 0, stream>>>(yb, h, ln2s, ln2b);
        run_gemm(stream, h, Dd,0,0, dec_ff_w1 + (long long)i*Dd*Ff, Ff,0,0,0, ff, Ff,0,0, dec_ff_b1 + (long long)i*Ff, nullptr,0,0,0, MB,(int)Ff,(int)Dd, 1.f,1, 1,1);
        run_gemm(stream, ff, Ff,0,0, dec_ff_w2 + (long long)i*Ff*Dd, Dd,0,0,0, yb, Dd,0,0, dec_ff_b2 + (long long)i*Dd, yb, Dd,0,0, MB,(int)Dd,(int)Ff, 1.f,0, 1,1);
    }

    // final norm + generator
    layernorm_kernel<<<MB, 256, 0, stream>>>(yb, h, dec_norm, dec_norm + Dd);
    run_gemm(stream, h, Dd,0,0, gen_w, Vv,0,0,0, (float*)d_out, Vv,0,0, gen_b, nullptr,0,0,0, MB,(int)Vv,(int)Dd, 1.f,0, 1,1);
}
